// VanillaRNN_33852932227208
// MI455X (gfx1250) — matrix operand order repaired and checked by reading, not compiled, not run on hardware
//
#include <hip/hip_runtime.h>

// ---------------------------------------------------------------------------
// VanillaRNN on gfx1250 (MI455X), wave32 + WMMA bf16 path.
//   h_{t+1} = tanh([h_t | x_t] @ [W; U] + bh)      (fused K = H + D = 640)
//   out     = h_T @ V + bp
// One persistent workgroup per 16 batch rows runs all T steps.
// Time-invariant weights are pinned on-chip: K[0,256) in VGPRs per wave,
// K[256,512) in a 256 KB LDS cache, only K[512,640) (the U part) streamed
// from L2 each step. 16 waves x 2 N-tiles: each A-fragment feeds 2 WMMAs.
// Next-step x tile prefetched via CDNA5 async global->LDS (ASYNCcnt).
// ---------------------------------------------------------------------------

typedef __attribute__((ext_vector_type(16))) __bf16 v16bf;
typedef __attribute__((ext_vector_type(8)))  float  v8f;

#define BB 512
#define TT 512
#define DD 128
#define HH 512
#define CC 256
#define KK (HH + DD)       // 640
#define KREG 256           // K-range held in per-wave VGPRs
#define KLDS 256           // K-range held in LDS weight cache
#define NW 16              // waves per workgroup
#define NTHR (NW * 32)     // 512 threads

__device__ __forceinline__ unsigned short f2bf(float f) {
    union { float f; unsigned u; } v; v.f = f;
    unsigned u = v.u;
    unsigned r = 0x7FFFu + ((u >> 16) & 1u);
    return (unsigned short)((u + r) >> 16);
}

__device__ __forceinline__ float fast_tanh(float v) {
#if __has_builtin(__builtin_amdgcn_tanhf)
    return __builtin_amdgcn_tanhf(v);          // V_TANH_F32
#elif __has_builtin(__builtin_amdgcn_tanh_f32)
    return __builtin_amdgcn_tanh_f32(v);
#else
    float e = __expf(2.0f * v);
    return 1.0f - 2.0f / (e + 1.0f);
#endif
}

// CDNA5 async global->LDS copy (ASYNCcnt). lds_off = low 32 bits of generic
// __shared__ pointer (flat->LDS aperture truncation, ISA 10.2).
__device__ __forceinline__ void async_copy_b128(unsigned lds_off,
                                                unsigned long long gaddr) {
    asm volatile("global_load_async_to_lds_b128 %0, %1, off"
                 :: "v"(lds_off), "v"(gaddr)
                 : "memory");
}
__device__ __forceinline__ void wait_async0() {
#if __has_builtin(__builtin_amdgcn_s_wait_asynccnt)
    __builtin_amdgcn_s_wait_asynccnt(0);
#else
    asm volatile("s_wait_asynccnt 0x0" ::: "memory");
#endif
}
__device__ __forceinline__ unsigned lds_off32(const void* p) {
    return (unsigned)(unsigned long long)p;
}

union Frag16 { v16bf v; uint4 q[2]; };

// ---------------------------------------------------------------------------
// Kernel 1: pack weights once (transposed; lane B-fragment = 32 contig bytes).
//   WcatT[n][k] bf16 512x640: k<512 -> W[k][n]; k>=512 -> U[k-512][n]
//   Vt[c][k]    bf16 256x512: V[k][c]
// ---------------------------------------------------------------------------
__global__ void pack_weights(const float* __restrict__ U,
                             const float* __restrict__ W,
                             const float* __restrict__ V,
                             unsigned short* __restrict__ WcatT,
                             unsigned short* __restrict__ Vt) {
    int i = blockIdx.x * blockDim.x + threadIdx.x;
    if (i < HH * KK) {
        int n = i / KK, k = i % KK;
        float v = (k < HH) ? W[k * HH + n] : U[(k - HH) * HH + n];
        WcatT[i] = f2bf(v);
    }
    if (i < CC * HH) {
        int c = i / HH, k = i % HH;
        Vt[i] = f2bf(V[k * CC + c]);
    }
}

__global__ void pack_x(const float* __restrict__ x,
                       unsigned short* __restrict__ xbf, int n) {
    int i = blockIdx.x * blockDim.x + threadIdx.x;
    if (i < n) xbf[i] = f2bf(x[i]);
}

// A-fragment (16x32 bf16) from an LDS row (K run = k0 + (lane<16 ? 0 : 16); M = lane%16).
__device__ __forceinline__ v16bf load_a_lds(const unsigned short* row,
                                            int k0, int lhalf) {
    Frag16 f;
    f.q[0] = *(const uint4*)(row + k0 + lhalf * 16);
    f.q[1] = *(const uint4*)(row + k0 + lhalf * 16 + 8);
    return f.v;
}

// B-fragment (32x16 bf16) from an [N][K]-major array (global or LDS);
// colbase points at this lane's column; K run = k0 + (lane<16 ? 0 : 16).
__device__ __forceinline__ v16bf load_b_col(const unsigned short* colbase,
                                            int k0, int lhalf) {
    const unsigned short* p = colbase + k0 + lhalf * 16;
    Frag16 f;
    f.q[0] = *(const uint4*)(p);
    f.q[1] = *(const uint4*)(p + 8);
    return f.v;
}

// ---------------------------------------------------------------------------
// Kernel 2: persistent scan. grid = B/16 = 32 workgroups, 512 threads each.
// LDS: zbuf 40 KB (double-buffered [h|x] state) + wcache 256 KB = 296 KB.
// ---------------------------------------------------------------------------
template <bool USE_ASYNC>
__global__ __launch_bounds__(NTHR)
void rnn_persist(const float* __restrict__ x,
                 const unsigned short* __restrict__ xbf,
                 const float* __restrict__ bh,
                 const float* __restrict__ bp,
                 const unsigned short* __restrict__ WcatT,
                 const unsigned short* __restrict__ Vt,
                 float* __restrict__ out) {
    __shared__ __attribute__((aligned(16))) unsigned short zbuf[2][16][KK];
    __shared__ __attribute__((aligned(16))) unsigned short wcache[HH][KLDS];

    const int tid   = threadIdx.x;
    const int lane  = tid & 31;
    const int wave  = tid >> 5;        // 0..15
    const int lmod  = lane & 15;
    const int lhalf = lane >> 4;
    const int b0    = blockIdx.x * 16;
    const int n0    = wave * 32;       // this wave's 2 adjacent N-tiles

    // ---- fill LDS weight cache: wcache[n][kk] = WcatT[n][KREG+kk] ----
    for (int i = tid; i < HH * (KLDS / 8); i += NTHR) {
        int n = i >> 5, c = i & 31;    // KLDS/8 = 32 chunks of 8 bf16
        *(uint4*)&wcache[n][c * 8] =
            *(const uint4*)&WcatT[(size_t)n * KK + KREG + c * 8];
    }

    // ---- init zbuf[0]: h = 0, x_{t=0} in K = 512..639 ----
    for (int i = tid; i < 16 * HH; i += NTHR) {
        zbuf[0][i / HH][i % HH] = 0;
    }
    if (USE_ASYNC) {
        if (tid < 256) {               // 16 rows x 16 chunks of 16 B
            int r = tid >> 4, c = tid & 15;
            async_copy_b128(
                lds_off32(&zbuf[0][r][HH + c * 8]),
                (unsigned long long)(xbf + ((size_t)(b0 + r) * TT + 0) * DD + c * 8));
        }
        wait_async0();
    } else {
        for (int i = tid; i < 16 * DD; i += NTHR) {
            int r = i / DD, k = i % DD;
            zbuf[0][r][HH + k] = f2bf(x[((b0 + r) * TT + 0) * DD + k]);
        }
    }
    __syncthreads();

    // ---- pin K[0,256) weight fragments in VGPRs (time-invariant) ----
    const unsigned short* wcol0 = WcatT + (size_t)(n0 + lmod) * KK;
    const unsigned short* wcol1 = WcatT + (size_t)(n0 + 16 + lmod) * KK;
    v16bf wreg[2][KREG / 32];          // 2 tiles x 8 frags = 128 VGPRs
#pragma unroll
    for (int j = 0; j < KREG / 32; ++j) {
        wreg[0][j] = load_b_col(wcol0, j * 32, lhalf);
        wreg[1][j] = load_b_col(wcol1, j * 32, lhalf);
    }

    const float bhv0 = bh[n0 + lmod];
    const float bhv1 = bh[n0 + 16 + lmod];
    const unsigned short* wc0 = &wcache[n0 + lmod][0];
    const unsigned short* wc1 = &wcache[n0 + 16 + lmod][0];

    for (int t = 0; t < TT; ++t) {
        const int p = t & 1;

        // async prefetch of x_{t+1} overlaps the whole step's GEMM
        if (USE_ASYNC && (t + 1 < TT) && tid < 256) {
            int r = tid >> 4, c = tid & 15;
            async_copy_b128(
                lds_off32(&zbuf[1 - p][r][HH + c * 8]),
                (unsigned long long)(xbf + ((size_t)(b0 + r) * TT + (t + 1)) * DD + c * 8));
        }

        v8f acc0 = {0.f, 0.f, 0.f, 0.f, 0.f, 0.f, 0.f, 0.f};
        v8f acc1 = {0.f, 0.f, 0.f, 0.f, 0.f, 0.f, 0.f, 0.f};
        const unsigned short* arow = &zbuf[p][lmod][0];

        // phase 1: K[0,256) — B from registers, A from LDS (1 frag : 2 WMMA)
#pragma unroll
        for (int j = 0; j < KREG / 32; ++j) {
            v16bf a = load_a_lds(arow, j * 32, lhalf);
            acc0 = __builtin_amdgcn_wmma_f32_16x16x32_bf16(
                false, a, false, wreg[0][j], (short)0, acc0, false, false);
            acc1 = __builtin_amdgcn_wmma_f32_16x16x32_bf16(
                false, a, false, wreg[1][j], (short)0, acc1, false, false);
        }
        // phase 2: K[256,512) — B from LDS weight cache
#pragma unroll
        for (int j = 0; j < KLDS / 32; ++j) {
            v16bf a  = load_a_lds(arow, KREG + j * 32, lhalf);
            v16bf b0 = load_b_col(wc0, j * 32, lhalf);
            v16bf b1 = load_b_col(wc1, j * 32, lhalf);
            acc0 = __builtin_amdgcn_wmma_f32_16x16x32_bf16(
                false, a, false, b0, (short)0, acc0, false, false);
            acc1 = __builtin_amdgcn_wmma_f32_16x16x32_bf16(
                false, a, false, b1, (short)0, acc1, false, false);
        }
        // phase 3: K[512,640) — U part streamed from L2 (128 KB/WG/step)
#pragma unroll
        for (int j = 0; j < (KK - HH) / 32; ++j) {
            v16bf a  = load_a_lds(arow, HH + j * 32, lhalf);
            v16bf b0 = load_b_col(wcol0, HH + j * 32, lhalf);
            v16bf b1 = load_b_col(wcol1, HH + j * 32, lhalf);
            acc0 = __builtin_amdgcn_wmma_f32_16x16x32_bf16(
                false, a, false, b0, (short)0, acc0, false, false);
            acc1 = __builtin_amdgcn_wmma_f32_16x16x32_bf16(
                false, a, false, b1, (short)0, acc1, false, false);
        }

        // h_new = tanh(acc + bh); scatter per C/D layout:
        // VGPR r, this lane -> (M = r + 8*lhalf, N = col)
#pragma unroll
        for (int r = 0; r < 8; ++r) {
            zbuf[1 - p][r + 8 * lhalf][n0 + lmod] =
                f2bf(fast_tanh(acc0[r] + bhv0));
            zbuf[1 - p][r + 8 * lhalf][n0 + 16 + lmod] =
                f2bf(fast_tanh(acc1[r] + bhv1));
        }

        if (USE_ASYNC) {
            wait_async0();             // x_{t+1} landed before the barrier
        } else if (t + 1 < TT) {
            for (int i = tid; i < 16 * DD; i += NTHR) {
                int r = i / DD, k = i % DD;
                zbuf[1 - p][r][HH + k] =
                    f2bf(x[((b0 + r) * TT + (t + 1)) * DD + k]);
            }
        }
        __syncthreads();
    }

    // ---- final projection: out = h_T @ V + bp; final h in zbuf[TT & 1] ----
    const int pf = TT & 1;
    {
        const int c0 = wave * 16;      // 16 waves x 16 cols = CC
        v8f acc = {0.f, 0.f, 0.f, 0.f, 0.f, 0.f, 0.f, 0.f};
        const unsigned short* arow = &zbuf[pf][lmod][0];
        const unsigned short* vcol = Vt + (size_t)(c0 + lmod) * HH;
#pragma unroll 4
        for (int k0 = 0; k0 < HH; k0 += 32) {
            v16bf a = load_a_lds(arow, k0, lhalf);
            v16bf b = load_b_col(vcol, k0, lhalf);
            acc = __builtin_amdgcn_wmma_f32_16x16x32_bf16(
                false, a, false, b, (short)0, acc, false, false);
        }
        const float bpv = bp[c0 + lmod];
#pragma unroll
        for (int r = 0; r < 8; ++r) {
            out[(b0 + r + 8 * lhalf) * CC + c0 + lmod] = acc[r] + bpv;
        }
    }
}

// ---------------------------------------------------------------------------
extern "C" void kernel_launch(void* const* d_in, const int* in_sizes, int n_in,
                              void* d_out, int out_size, void* d_ws, size_t ws_size,
                              hipStream_t stream) {
    const float* x  = (const float*)d_in[0];
    const float* U  = (const float*)d_in[1];
    const float* W  = (const float*)d_in[2];
    const float* V  = (const float*)d_in[3];
    const float* bh = (const float*)d_in[4];
    const float* bp = (const float*)d_in[5];
    float* out = (float*)d_out;

    const size_t wbytes = (size_t)(HH * KK + CC * HH) * 2;   // 917504 B
    const size_t xbytes = (size_t)BB * TT * DD * 2;          // 64 MB bf16 x

    unsigned short* WcatT = (unsigned short*)d_ws;
    unsigned short* Vt    = (unsigned short*)((char*)d_ws + (size_t)HH * KK * 2);

    pack_weights<<<(HH * KK + 255) / 256, 256, 0, stream>>>(U, W, V, WcatT, Vt);

    if (ws_size >= wbytes + xbytes) {
        unsigned short* xbf = (unsigned short*)((char*)d_ws + wbytes);
        const int nx = BB * TT * DD;
        pack_x<<<(nx + 255) / 256, 256, 0, stream>>>(x, xbf, nx);
        rnn_persist<true><<<BB / 16, NTHR, 0, stream>>>(x, xbf, bh, bp,
                                                        WcatT, Vt, out);
    } else {
        rnn_persist<false><<<BB / 16, NTHR, 0, stream>>>(x, nullptr, bh, bp,
                                                         WcatT, Vt, out);
    }
}
